// SpiralDeblock_45810121179172
// MI455X (gfx1250) — compile-verified, matrix operand order checked by reading, not run on hardware
//
#include <hip/hip_runtime.h>
#include <hip/hip_bf16.h>

typedef __attribute__((ext_vector_type(16))) __bf16 v16bf;
typedef __attribute__((ext_vector_type(8)))  __bf16 v8bf;
typedef __attribute__((ext_vector_type(8)))  float  v8f;
typedef __attribute__((ext_vector_type(4)))  float  v4f;

#define B_      16
#define N_IN    6250
#define N_OUT   25000
#define C_      64
#define S_      9
#define NNZ_    75000
#define K_TOT   576          // S_*C_
#define KSTEPS  18           // K_TOT/32
#define NT_PER_B 1563        // ceil(N_OUT/16)
#define TILES   (B_ * NT_PER_B)

// ---------------------------------------------------------------------------
// Stage 1: scatter-add pooling. pooled[b,row,:] += x[b,col,:] * val
// One wave per (b, nnz) pair; each lane covers 2 channels (float2) and
// issues two non-returning global_atomic_add_f32 (STOREcnt path).
// ---------------------------------------------------------------------------
__global__ __launch_bounds__(256) void sd_scatter_pool(
    const float* __restrict__ x, const float* __restrict__ vals,
    const int* __restrict__ rows, const int* __restrict__ cols,
    float* __restrict__ pooled) {
  long long tid = (long long)blockIdx.x * blockDim.x + threadIdx.x;
  long long total = (long long)NNZ_ * B_ * 32;
  if (tid >= total) return;
  int lane = (int)(tid & 31);
  long long pair = tid >> 5;
  int i = (int)(pair % NNZ_);
  int b = (int)(pair / NNZ_);
  float v  = vals[i];
  int col  = cols[i];
  int row  = rows[i];
  const float2* xp = (const float2*)(x + ((size_t)b * N_IN + col) * C_);
  float2 xv = xp[lane];
  float* dst = pooled + ((size_t)b * N_OUT + row) * C_ + lane * 2;
  unsafeAtomicAdd(dst,     xv.x * v);
  unsafeAtomicAdd(dst + 1, xv.y * v);
}

// ---------------------------------------------------------------------------
// Stage 2: f32 pooled -> bf16 pooled. The f32 buffer is dead afterwards:
// read it non-temporally so it doesn't displace anything in L2. The bf16
// result (51 MB) is the hot gather working set -> regular-temporal store.
// ---------------------------------------------------------------------------
__global__ __launch_bounds__(256) void sd_cvt_bf16(
    const float* __restrict__ src, __bf16* __restrict__ dst, int n8) {
  int t = blockIdx.x * blockDim.x + threadIdx.x;
  if (t >= n8) return;
  const v4f* s = (const v4f*)src;
  v4f a = __builtin_nontemporal_load(&s[2 * t]);
  v4f b = __builtin_nontemporal_load(&s[2 * t + 1]);
  v8bf o;
  o[0] = (__bf16)a[0]; o[1] = (__bf16)a[1]; o[2] = (__bf16)a[2]; o[3] = (__bf16)a[3];
  o[4] = (__bf16)b[0]; o[5] = (__bf16)b[1]; o[6] = (__bf16)b[2]; o[7] = (__bf16)b[3];
  ((v8bf*)dst)[t] = o;
}

// ---------------------------------------------------------------------------
// Stage 3: shuffle W (576x64 f32, row-major) into fragment-ready bf16.
// Fragment f = kstep*4 + ntile is 512 bf16: lane L holds 16 elems at
// f*512 + L*16. Element j of lane L is W[k, n] with
//   n = ntile*16 + (L & 15)
//   k = kstep*32 + (j<8 ? j : j+8) + ((L>=16) ? 8 : 0)
// matching the ISA 16-bit B 32x16 per-lane K split (symmetric to A layout).
// ---------------------------------------------------------------------------
__global__ __launch_bounds__(256) void sd_shuffle_w(
    const float* __restrict__ W, __bf16* __restrict__ wshuf) {
  int tid = blockIdx.x * blockDim.x + threadIdx.x;
  if (tid >= KSTEPS * 4 * 512) return;
  int frag = tid >> 9;
  int r    = tid & 511;
  int lane = r >> 4;
  int j    = r & 15;
  int ks   = frag >> 2;
  int t    = frag & 3;
  int kl   = (j < 8 ? j : j + 8) + ((lane >= 16) ? 8 : 0);
  int k    = ks * 32 + kl;
  int n    = t * 16 + (lane & 15);
  wshuf[tid] = (__bf16)W[k * 64 + n];
}

// ---------------------------------------------------------------------------
// Stage 4: spiral gather + GEMM + bias + ELU, one wave per 16-row tile.
// A fragments (16x32 bf16) loaded straight from bf16 pooled via hoisted
// per-segment row pointers (ISA 16-bit A layout: lane 0-15 = K {b..b+7,
// b+16..b+23}, lanes 16-31 the +8 halves -> two 16B loads/lane/kstep).
// B fragments: one coalesced 32B/lane load from pre-shuffled W.
// Output streams out with NT stores so the L2 keeps the gather set.
// ---------------------------------------------------------------------------
__global__ __launch_bounds__(256) void sd_spiral_gemm(
    const __bf16* __restrict__ pooled, const __bf16* __restrict__ wshuf,
    const int* __restrict__ spiral, const float* __restrict__ bias,
    float* __restrict__ out) {
  int wave = (int)(((long long)blockIdx.x * blockDim.x + threadIdx.x) >> 5);
  if (wave >= TILES) return;
  int lane = threadIdx.x & 31;
  int b  = wave / NT_PER_B;
  int nt = wave - b * NT_PER_B;
  int m    = lane & 15;
  int half = lane >> 4;
  int n = nt * 16 + m;
  if (n >= N_OUT) n = N_OUT - 1;          // clamp loads; stores are guarded

  const int* sp = spiral + n * S_;
  const __bf16* pb = pooled + (size_t)b * N_OUT * C_ + half * 8;

  // Hoist the 9 gathered row pointers (per-lane) out of the K loop.
  const __bf16* rowp[S_];
#pragma unroll
  for (int s = 0; s < S_; ++s) rowp[s] = pb + (size_t)sp[s] * C_;

  v8f acc[4] = {};
  const v16bf* wb = (const v16bf*)wshuf;

#pragma unroll
  for (int ks = 0; ks < KSTEPS; ++ks) {
    const int seg = ks >> 1;              // which spiral neighbor (64 ch each)
    const int off = (ks & 1) * 32;        // half of the 64-ch segment
    const __bf16* rp = rowp[seg] + off;
    v8bf alo = *(const v8bf*)rp;          // K = base+h8 .. +7
    v8bf ahi = *(const v8bf*)(rp + 16);   // K = base+h8+16 .. +23
    v16bf afrag = __builtin_shufflevector(
        alo, ahi, 0, 1, 2, 3, 4, 5, 6, 7, 8, 9, 10, 11, 12, 13, 14, 15);
#pragma unroll
    for (int t = 0; t < 4; ++t) {
      v16bf bfrag = wb[(ks * 4 + t) * 32 + lane];
      acc[t] = __builtin_amdgcn_wmma_f32_16x16x32_bf16(
          false, afrag, false, bfrag, (short)0, acc[t], false, false);
    }
  }

  // Epilogue: C/D layout -> VGPR g: lanes 0-15 = (M=g, N=lane),
  // lanes 16-31 = (M=g+8, N=lane-16). Bias + ELU, NT streaming stores.
  const size_t obase = (size_t)b * N_OUT * C_;
#pragma unroll
  for (int t = 0; t < 4; ++t) {
    int c = t * 16 + m;
    float bc = bias[c];
#pragma unroll
    for (int g = 0; g < 8; ++g) {
      int mr = g + half * 8;
      int nn = nt * 16 + mr;
      if (nn < N_OUT) {
        float v = acc[t][g] + bc;
        float r = (v > 0.0f) ? v : expm1f(v);
        __builtin_nontemporal_store(r, &out[obase + (size_t)nn * C_ + c]);
      }
    }
  }
}

// ---------------------------------------------------------------------------
extern "C" void kernel_launch(void* const* d_in, const int* in_sizes, int n_in,
                              void* d_out, int out_size, void* d_ws, size_t ws_size,
                              hipStream_t stream) {
  const float* x      = (const float*)d_in[0];
  const float* tvals  = (const float*)d_in[1];
  const int*   trow   = (const int*)d_in[2];
  const int*   tcol   = (const int*)d_in[3];
  const int*   spiral = (const int*)d_in[4];
  const float* W      = (const float*)d_in[5];
  const float* bias   = (const float*)d_in[6];
  float* out = (float*)d_out;

  char* ws = (char*)d_ws;
  const size_t pooled_elems = (size_t)B_ * N_OUT * C_;        // 25.6M
  float*  pooled_f32 = (float*)ws;                            // 102,400,000 B
  __bf16* pooled_bf  = (__bf16*)(ws + 102400000);             //  51,200,000 B
  __bf16* wshuf      = (__bf16*)(ws + 102400000 + 51200128);  //      73,728 B

  (void)hipMemsetAsync(pooled_f32, 0, pooled_elems * sizeof(float), stream);

  long long sthreads = (long long)NNZ_ * B_ * 32;
  sd_scatter_pool<<<(int)((sthreads + 255) / 256), 256, 0, stream>>>(
      x, tvals, trow, tcol, pooled_f32);

  int n8 = (int)(pooled_elems / 8);
  sd_cvt_bf16<<<(n8 + 255) / 256, 256, 0, stream>>>(pooled_f32, pooled_bf, n8);

  sd_shuffle_w<<<(KSTEPS * 4 * 512 + 255) / 256, 256, 0, stream>>>(W, wshuf);

  int blocks = (TILES * 32 + 255) / 256;   // 3126 blocks, 8 waves each
  sd_spiral_gemm<<<blocks, 256, 0, stream>>>(pooled_bf, wshuf, spiral, bias, out);
}
